// MSRAMIL_27530740367377
// MI455X (gfx1250) — matrix operand order repaired and checked by reading, not compiled, not run on hardware
//
#include <hip/hip_runtime.h>

// ---------------------------------------------------------------------------
// CDNA5 (gfx1250) implementation of the TransMIL-like reference:
//   PPEG depthwise convs -> fc1 (WMMA) -> 2x criss-cross attention (WMMA
//   energy + strided softmax + WMMA map) -> 1-logit fc -> global softmax.
// All dense contractions run on v_wmma_f32_16x16x32_f16, f16 intermediates.
// GEMMs compute a 16x64 strip per wave (A-fragment reused over 4 B-frags).
// Workspace requirement: ~490 MB (regions aliased in time).
// ---------------------------------------------------------------------------

typedef __attribute__((ext_vector_type(16))) _Float16 v16h;
typedef __attribute__((ext_vector_type(8)))  _Float16 v8h;
typedef __attribute__((ext_vector_type(8)))  float    v8f;

union V16 { v16h v; v8h h[2]; };

__device__ __forceinline__ v8f wmma32(v16h a, v16h b, v8f c) {
    return __builtin_amdgcn_wmma_f32_16x16x32_f16(false, a, false, b,
                                                  (short)0, c, false, false);
}

// A fragment: 16x32 f16, row-major source, lda in halves.
// lane m = lane&15, kb = 16*(lane>=16); halves 0..15 = K kb..kb+15
// (contiguous, matching load_frag_b).
__device__ __forceinline__ v16h load_frag_a(const _Float16* __restrict__ A, int lda) {
    int lane = threadIdx.x & 31;
    int m = lane & 15;
    int kb = (lane >> 4) * 16;
    const _Float16* r = A + (size_t)m * lda + kb;
    V16 u;
    u.h[0] = *(const v8h*)(r);
    u.h[1] = *(const v8h*)(r + 8);
    return u.v;
}

// B fragment: B is 32x16; loaded from Bt = B^T row-major (N x K), ldb in halves.
// lane n = lane&15 holds 16 contiguous K at kb = 16*(lane>=16)
// (same contiguous K order as load_frag_a).
__device__ __forceinline__ v16h load_frag_b(const _Float16* __restrict__ Bt, int ldb) {
    int lane = threadIdx.x & 31;
    int n = lane & 15;
    int kb = (lane >> 4) * 16;
    const _Float16* r = Bt + (size_t)n * ldb + kb;
    V16 u;
    u.h[0] = *(const v8h*)(r);
    u.h[1] = *(const v8h*)(r + 8);
    return u.v;
}

// D 16x16 f32: VGPR r holds M = r + 8*(lane>=16), N = lane&15.
__device__ __forceinline__ void store_d_f32(float* __restrict__ C, int ldc, v8f d) {
    int lane = threadIdx.x & 31;
    int n = lane & 15;
    int mb = (lane >> 4) * 8;
#pragma unroll
    for (int r = 0; r < 8; ++r) C[(size_t)(mb + r) * ldc + n] = d[r];
}

// ---------------------------------------------------------------------------
// f32 -> f16 conversion (weights)
// ---------------------------------------------------------------------------
__global__ __launch_bounds__(256) void f32_to_f16_kernel(
    const float* __restrict__ s, _Float16* __restrict__ d, int n) {
    int i = blockIdx.x * 256 + threadIdx.x;
    if (i < n) d[i] = (_Float16)s[i];
}

// ---------------------------------------------------------------------------
// PPEG: h1 = cnn + dw7(cnn)+b7 + dw5(cnn)+b5 + dw3(cnn)+b3, output f16
// h0: [65536,1024] f32 pixel-major; block = 16x16 spatial tile x 8 channels.
// ---------------------------------------------------------------------------
__global__ __launch_bounds__(256) void ppeg_kernel(
    const float* __restrict__ h0,
    const float* __restrict__ w7, const float* __restrict__ b7,
    const float* __restrict__ w5, const float* __restrict__ b5,
    const float* __restrict__ w3, const float* __restrict__ b3,
    _Float16* __restrict__ h1) {
    __shared__ float tile[22 * 22 * 8];
    __shared__ float wsm[8 * 83];
    __shared__ float cb[8];
    int tx0 = blockIdx.x * 16, ty0 = blockIdx.y * 16, c0 = blockIdx.z * 8;
    int tid = threadIdx.x;

    for (int i = tid; i < 8 * 83; i += 256) {
        int c = i / 83, t = i % 83;
        float w;
        if (t < 49)      w = w7[(size_t)(c0 + c) * 49 + t];
        else if (t < 74) w = w5[(size_t)(c0 + c) * 25 + (t - 49)];
        else             w = w3[(size_t)(c0 + c) * 9 + (t - 74)];
        wsm[c * 83 + t] = w;
    }
    if (tid < 8) cb[tid] = b7[c0 + tid] + b5[c0 + tid] + b3[c0 + tid];

    for (int i = tid; i < 22 * 22 * 8; i += 256) {
        int c = i & 7, p = i >> 3;
        int yy = ty0 - 3 + p / 22, xx = tx0 - 3 + p % 22;
        float vv = 0.f;
        if (yy >= 0 && yy < 256 && xx >= 0 && xx < 256)
            vv = h0[((size_t)yy * 256 + xx) * 1024 + c0 + c];
        tile[p * 8 + c] = vv;
    }
    __syncthreads();

    for (int r = 0; r < 8; ++r) {
        int idx = tid + r * 256;
        int c = idx & 7, p = idx >> 3;
        int py = p >> 4, px = p & 15;
        const float* wp = &wsm[c * 83];
        float acc = tile[((py + 3) * 22 + px + 3) * 8 + c] + cb[c];
#pragma unroll
        for (int dy = 0; dy < 7; ++dy)
#pragma unroll
            for (int dx = 0; dx < 7; ++dx)
                acc += tile[((py + dy) * 22 + px + dx) * 8 + c] * wp[dy * 7 + dx];
#pragma unroll
        for (int dy = 0; dy < 5; ++dy)
#pragma unroll
            for (int dx = 0; dx < 5; ++dx)
                acc += tile[((py + 1 + dy) * 22 + px + 1 + dx) * 8 + c] * wp[49 + dy * 5 + dx];
#pragma unroll
        for (int dy = 0; dy < 3; ++dy)
#pragma unroll
            for (int dx = 0; dx < 3; ++dx)
                acc += tile[((py + 2 + dy) * 22 + px + 2 + dx) * 8 + c] * wp[74 + dy * 3 + dx];
        h1[((size_t)(ty0 + py) * 256 + tx0 + px) * 1024 + c0 + c] = (_Float16)acc;
    }
}

// ---------------------------------------------------------------------------
// Generic GEMM: Y[M,N] = act( X[M,K] @ W[N,K]^T + bias ), f16 in/out, f32 acc.
// One wave per 16x64 output strip (4 accumulators, A fragment reused 4x).
// Requires Nout % 64 == 0.
// ---------------------------------------------------------------------------
__global__ __launch_bounds__(256) void gemm_bias_act(
    const _Float16* __restrict__ X, const _Float16* __restrict__ W,
    const float* __restrict__ bias, _Float16* __restrict__ Y,
    int M, int K, int Nout, int relu) {
    int widx = blockIdx.x * 8 + (threadIdx.x >> 5);
    int ngroups = Nout >> 6;                   // groups of 4 n-tiles
    int ng = widx % ngroups, mt = widx / ngroups;
    const _Float16* A0 = X + (size_t)(mt * 16) * K;
    const _Float16* B0 = W + (size_t)(ng * 64) * K;
    v8f acc[4] = {};
    for (int k0 = 0; k0 < K; k0 += 32) {
        if (k0 + 64 < K) {
            __builtin_prefetch(A0 + k0 + 64, 0, 3);   // global_prefetch_b8 (WGP)
            __builtin_prefetch(B0 + k0 + 64, 0, 3);
        }
        v16h a = load_frag_a(A0 + k0, K);
#pragma unroll
        for (int j = 0; j < 4; ++j) {
            v16h b = load_frag_b(B0 + (size_t)(j * 16) * K + k0, K);
            acc[j] = wmma32(a, b, acc[j]);
        }
    }
    int lane = threadIdx.x & 31;
    int n = lane & 15, mb = (lane >> 4) * 8;
#pragma unroll
    for (int j = 0; j < 4; ++j) {
        float bv = bias[ng * 64 + j * 16 + n];
        _Float16* Yp = Y + (size_t)(mt * 16) * Nout + ng * 64 + j * 16;
#pragma unroll
        for (int r = 0; r < 8; ++r) {
            float d = acc[j][r] + bv;
            if (relu) d = fmaxf(d, 0.f);
            Yp[(size_t)(mb + r) * Nout + n] = (_Float16)d;
        }
    }
}

// ---------------------------------------------------------------------------
// Criss-cross energy: E[bx][r][n] = sum_c q[pix(r)][c] * k[pix(n)][c],
// pix(i) = bx*baseMul + i*pixStride. K = 64 (two WMMA steps).
// EH: baseMul=1,pixStride=256 (per column w). EW: baseMul=256,pixStride=1.
// ---------------------------------------------------------------------------
__global__ __launch_bounds__(256) void cc_energy(
    const _Float16* __restrict__ q, const _Float16* __restrict__ k,
    float* __restrict__ E, int baseMul, int pixStride) {
    int bx = blockIdx.x;
    int mt = blockIdx.y;
    int nt = blockIdx.z * 8 + (threadIdx.x >> 5);
    size_t base = (size_t)bx * baseMul;
    const _Float16* A0 = q + (base + (size_t)mt * 16 * pixStride) * 64;
    const _Float16* B0 = k + (base + (size_t)nt * 16 * pixStride) * 64;
    int lda = pixStride * 64;
    v8f acc = {};
    acc = wmma32(load_frag_a(A0, lda), load_frag_b(B0, lda), acc);
    acc = wmma32(load_frag_a(A0 + 32, lda), load_frag_b(B0 + 32, lda), acc);
    store_d_f32(E + (size_t)bx * 65536 + (size_t)mt * 16 * 256 + nt * 16, 256, acc);
}

// ---------------------------------------------------------------------------
// Softmax over the h axis (reference: softmax(energy, axis=1)).
// Element (h, w=bx, t): E[h*sH + bx*sW + t]. Online max/sum, then write f16.
// ---------------------------------------------------------------------------
__global__ __launch_bounds__(256) void cc_smax_stat(
    const float* __restrict__ E, float* __restrict__ mOut,
    float* __restrict__ sOut, int sH, int sW) {
    int bx = blockIdx.x, t = threadIdx.x;
    const float* p = E + (size_t)bx * sW + t;
    float m = -3.4e38f, s = 0.f;
    for (int h = 0; h < 256; ++h) {
        float x = p[(size_t)h * sH];
        float nm = fmaxf(m, x);
        s = s * __expf(m - nm) + __expf(x - nm);
        m = nm;
    }
    mOut[bx * 256 + t] = m;
    sOut[bx * 256 + t] = s;
}

__global__ __launch_bounds__(256) void cc_smax_write(
    const float* __restrict__ E, const float* __restrict__ mIn,
    const float* __restrict__ sIn, _Float16* __restrict__ att, int sH, int sW) {
    int bx = blockIdx.x, t = threadIdx.x;
    float m = mIn[bx * 256 + t];
    float inv = 1.f / sIn[bx * 256 + t];
    const float* p = E + (size_t)bx * sW + t;
    _Float16* o = att + (size_t)bx * sW + t;
    for (int h = 0; h < 256; ++h)
        o[(size_t)h * sH] = (_Float16)(__expf(p[(size_t)h * sH] - m) * inv);
}

// ---------------------------------------------------------------------------
// Criss-cross map: D[r][c] = sum_g att[bx][r][g] * v[pix(g)][c].
// mode 0: write f32 tmp. mode 1: y = gamma*(tmp + D) + x (f16).
// V tiles are LDS-transposed so B fragments are ds_load_b128 contiguous.
// Block: 64(r) x 64(c) output tile, 8 waves (mi = wid&3, nj = 2*(wid>>2)+t).
// ---------------------------------------------------------------------------
__global__ __launch_bounds__(256) void cc_map(
    const _Float16* __restrict__ att, const _Float16* __restrict__ vmat,
    const _Float16* __restrict__ xres, float* __restrict__ tmp,
    _Float16* __restrict__ y, const float* __restrict__ gamma,
    int baseMul, int pixStride, int mode) {
    __shared__ _Float16 ldsB[64 * 32];
    int bx = blockIdx.x;
    int r0 = blockIdx.y * 64;
    int c0 = blockIdx.z * 64;
    int tid = threadIdx.x, lane = tid & 31, wid = tid >> 5;
    int mi = wid & 3, njb = (wid >> 2) * 2;
    size_t base = (size_t)bx * baseMul;
    const _Float16* attRow = att + (size_t)bx * 65536 + (size_t)(r0 + mi * 16) * 256;
    v8f acc0 = {}, acc1 = {};
    for (int k0 = 0; k0 < 256; k0 += 32) {
        // stage V[k0..k0+31][c0..c0+63] transposed into LDS [c][k]
        int g = k0 + (tid >> 3);
        int cl = (tid & 7) * 8;
        const _Float16* src = vmat + (base + (size_t)g * pixStride) * 512 + c0 + cl;
        v8h val = *(const v8h*)src;
#pragma unroll
        for (int j = 0; j < 8; ++j) ldsB[(cl + j) * 32 + (tid >> 3)] = val[j];
        __syncthreads();

        v16h a = load_frag_a(attRow + k0, 256);
        int n = lane & 15, kb = (lane >> 4) * 16;
        const _Float16* b0p = &ldsB[(njb * 16 + n) * 32 + kb];
        const _Float16* b1p = &ldsB[((njb + 1) * 16 + n) * 32 + kb];
        V16 ub0, ub1;
        ub0.h[0] = *(const v8h*)b0p; ub0.h[1] = *(const v8h*)(b0p + 8);
        ub1.h[0] = *(const v8h*)b1p; ub1.h[1] = *(const v8h*)(b1p + 8);
        acc0 = wmma32(a, ub0.v, acc0);
        acc1 = wmma32(a, ub1.v, acc1);
        __syncthreads();
    }
    int n = lane & 15, mb = (lane >> 4) * 8;
    float gm = mode ? gamma[0] : 0.f;
#pragma unroll
    for (int r = 0; r < 8; ++r) {
        int rr = r0 + mi * 16 + mb + r;
        size_t pix = base + (size_t)rr * pixStride;
#pragma unroll
        for (int t = 0; t < 2; ++t) {
            float d = t ? acc1[r] : acc0[r];
            size_t idx = pix * 512 + c0 + (njb + t) * 16 + n;
            if (mode == 0) {
                tmp[idx] = d;
            } else {
                float outv = gm * (tmp[idx] + d) + (float)xres[idx];
                y[idx] = (_Float16)outv;
            }
        }
    }
}

// ---------------------------------------------------------------------------
// logits[p] = dot(y2[p,:512], Wfc) + bfc  (one wave per pixel)
// ---------------------------------------------------------------------------
__global__ __launch_bounds__(256) void logits_kernel(
    const _Float16* __restrict__ x, const float* __restrict__ Wfc,
    const float* __restrict__ bfc, float* __restrict__ out) {
    int wid = threadIdx.x >> 5, lane = threadIdx.x & 31;
    int p = blockIdx.x * 8 + wid;
    const _Float16* row = x + (size_t)p * 512;
    float acc = 0.f;
#pragma unroll
    for (int i = 0; i < 16; ++i)
        acc += (float)row[lane + 32 * i] * Wfc[lane + 32 * i];
#pragma unroll
    for (int off = 16; off > 0; off >>= 1) acc += __shfl_xor(acc, off, 32);
    if (lane == 0) out[p] = acc + bfc[0];
}

// ---------------------------------------------------------------------------
// Global softmax over 65536 logits, single workgroup (deterministic).
// ---------------------------------------------------------------------------
__global__ __launch_bounds__(1024) void softmax_all(
    const float* __restrict__ logits, float* __restrict__ out) {
    __shared__ float red[1024];
    int tid = threadIdx.x;
    float m = -3.4e38f;
    for (int i = tid; i < 65536; i += 1024) m = fmaxf(m, logits[i]);
    red[tid] = m; __syncthreads();
    for (int s = 512; s > 0; s >>= 1) {
        if (tid < s) red[tid] = fmaxf(red[tid], red[tid + s]);
        __syncthreads();
    }
    m = red[0]; __syncthreads();
    float sum = 0.f;
    for (int i = tid; i < 65536; i += 1024) sum += __expf(logits[i] - m);
    red[tid] = sum; __syncthreads();
    for (int s = 512; s > 0; s >>= 1) {
        if (tid < s) red[tid] += red[tid + s];
        __syncthreads();
    }
    float inv = 1.f / red[0];
    for (int i = tid; i < 65536; i += 1024) out[i] = __expf(logits[i] - m) * inv;
}

// ---------------------------------------------------------------------------
extern "C" void kernel_launch(void* const* d_in, const int* in_sizes, int n_in,
                              void* d_out, int out_size, void* d_ws, size_t ws_size,
                              hipStream_t stream) {
    (void)in_sizes; (void)n_in; (void)out_size; (void)ws_size;
    const float* h0    = (const float*)d_in[0];
    const float* w7    = (const float*)d_in[1];
    const float* b7    = (const float*)d_in[2];
    const float* w5    = (const float*)d_in[3];
    const float* b5    = (const float*)d_in[4];
    const float* w3    = (const float*)d_in[5];
    const float* b3    = (const float*)d_in[6];
    const float* W1    = (const float*)d_in[7];
    const float* b1    = (const float*)d_in[8];
    const float* Wq    = (const float*)d_in[9];
    const float* bq    = (const float*)d_in[10];
    const float* Wk    = (const float*)d_in[11];
    const float* bk    = (const float*)d_in[12];
    const float* Wv    = (const float*)d_in[13];
    const float* bv    = (const float*)d_in[14];
    const float* gamma = (const float*)d_in[15];
    const float* Wq1   = (const float*)d_in[16];
    const float* bq1   = (const float*)d_in[17];
    const float* Wk1   = (const float*)d_in[18];
    const float* bk1   = (const float*)d_in[19];
    const float* Wv1   = (const float*)d_in[20];
    const float* bv1   = (const float*)d_in[21];
    const float* gamma1= (const float*)d_in[22];
    const float* Wfc   = (const float*)d_in[23];
    const float* bfc   = (const float*)d_in[24];

    // -------- workspace layout (regions aliased in time; ~490 MB total) -----
    char* p = (char*)d_ws;
    auto alloc = [&](size_t bytes) -> char* {
        char* r = p; p += (bytes + 255) & ~(size_t)255; return r;
    };
    char* R1 = alloc(134217728);             // h1f16 -> [EH|EW] -> outTmp
    _Float16* h1f    = (_Float16*)R1;
    float*    E0     = (float*)R1;
    float*    E1     = (float*)(R1 + 67108864);
    float*    outTmp = (float*)R1;
    _Float16* h2  = (_Float16*)alloc(67108864);
    _Float16* y1  = (_Float16*)alloc(67108864);
    _Float16* y2  = (_Float16*)alloc(67108864);
    _Float16* qb  = (_Float16*)alloc(8388608);
    _Float16* kb2 = (_Float16*)alloc(8388608);
    _Float16* vb  = (_Float16*)alloc(67108864);
    _Float16* attH = (_Float16*)alloc(33554432);
    _Float16* attW = (_Float16*)alloc(33554432);
    _Float16* W1h  = (_Float16*)alloc(512 * 1024 * 2);
    _Float16* Wqh  = (_Float16*)alloc(64 * 512 * 2);
    _Float16* Wkh  = (_Float16*)alloc(64 * 512 * 2);
    _Float16* Wvh  = (_Float16*)alloc(512 * 512 * 2);
    _Float16* Wq1h = (_Float16*)alloc(64 * 512 * 2);
    _Float16* Wk1h = (_Float16*)alloc(64 * 512 * 2);
    _Float16* Wv1h = (_Float16*)alloc(512 * 512 * 2);
    float* auxm = (float*)alloc(262144);
    float* auxs = (float*)alloc(262144);
    float* logitsBuf = (float*)alloc(262144);

    auto cvt = [&](const float* s, _Float16* d, int n) {
        f32_to_f16_kernel<<<(n + 255) / 256, 256, 0, stream>>>(s, d, n);
    };
    cvt(W1, W1h, 512 * 1024);
    cvt(Wq, Wqh, 64 * 512);   cvt(Wk, Wkh, 64 * 512);   cvt(Wv, Wvh, 512 * 512);
    cvt(Wq1, Wq1h, 64 * 512); cvt(Wk1, Wk1h, 64 * 512); cvt(Wv1, Wv1h, 512 * 512);

    // PPEG positional encoding -> h1 (f16)
    ppeg_kernel<<<dim3(16, 16, 128), 256, 0, stream>>>(h0, w7, b7, w5, b5, w3, b3, h1f);

    // fc1: [65536,1024]@[1024,512]^T + ReLU -> h2 (f16)
    // waves = (M/16)*(Nout/64) = 4096*8 -> 4096 blocks
    gemm_bias_act<<<4096, 256, 0, stream>>>(h1f, W1h, b1, h2, 65536, 1024, 512, 1);

    auto ccblock = [&](const _Float16* x, const _Float16* Wq_, const float* bq_,
                       const _Float16* Wk_, const float* bk_,
                       const _Float16* Wv_, const float* bv_,
                       const float* gm, _Float16* yout) {
        // q,k: waves = 4096*1 -> 512 blocks ; v: waves = 4096*8 -> 4096 blocks
        gemm_bias_act<<<512, 256, 0, stream>>>(x, Wq_, bq_, qb, 65536, 512, 64, 0);
        gemm_bias_act<<<512, 256, 0, stream>>>(x, Wk_, bk_, kb2, 65536, 512, 64, 0);
        gemm_bias_act<<<4096, 256, 0, stream>>>(x, Wv_, bv_, vb, 65536, 512, 512, 0);
        // energies: EH per column w, EW per row h
        cc_energy<<<dim3(256, 16, 2), 256, 0, stream>>>(qb, kb2, E0, 1, 256);
        cc_energy<<<dim3(256, 16, 2), 256, 0, stream>>>(qb, kb2, E1, 256, 1);
        // softmax over h axis (reference softmax(energy, 1))
        cc_smax_stat <<<256, 256, 0, stream>>>(E0, auxm, auxs, 256, 65536);
        cc_smax_write<<<256, 256, 0, stream>>>(E0, auxm, auxs, attH, 256, 65536);
        cc_smax_stat <<<256, 256, 0, stream>>>(E1, auxm, auxs, 65536, 256);
        cc_smax_write<<<256, 256, 0, stream>>>(E1, auxm, auxs, attW, 65536, 256);
        // map GEMMs: outH to tmp, then outW fused with gamma*(.)+x -> yout
        cc_map<<<dim3(256, 4, 8), 256, 0, stream>>>(attH, vb, nullptr, outTmp,
                                                    nullptr, gm, 1, 256, 0);
        cc_map<<<dim3(256, 4, 8), 256, 0, stream>>>(attW, vb, x, outTmp,
                                                    yout, gm, 256, 1, 1);
    };
    ccblock(h2, Wqh, bq, Wkh, bk, Wvh, bv, gamma, y1);
    ccblock(y1, Wq1h, bq1, Wk1h, bk1, Wv1h, bv1, gamma1, y2);

    logits_kernel<<<8192, 256, 0, stream>>>(y2, Wfc, bfc, logitsBuf);
    softmax_all<<<1, 1024, 0, stream>>>(logitsBuf, (float*)d_out);
}
